// KAN_3607772529058
// MI455X (gfx1250) — matrix operand order repaired and checked by reading, not compiled, not run on hardware
//
#include <hip/hip_runtime.h>

// ---------------------------------------------------------------------------
// KAN (Jacobi) layer as a bf16 WMMA GEMM on gfx1250.
//   y[b,o] = sum_{i,d} J_d(tanh(x[b,i])) * C[i,o,d]
// GEMM view: M=262144, K=1152 (k = i*9+d), N=128.
// Per K-section both operands are staged in LDS: A fragments are computed
// in-place (tanh + Jacobi recurrence), B fragments are copied once per
// workgroup from the L2-resident packed weights. Phase B software-pipelines
// the B fragment stream (2-deep rotation) against the WMMA pairs.
// ---------------------------------------------------------------------------

typedef __attribute__((ext_vector_type(16))) __bf16 v16bf;
typedef __attribute__((ext_vector_type(8)))  float  v8f;

#define IN_DIM    128
#define OUT_DIM   128
#define NDEG      9                       // DEGREE + 1
#define K_TOTAL   (IN_DIM * NDEG)         // 1152
#define KCHUNKS   (K_TOTAL / 32)          // 36 chunks of K=32
#define SECTIONS  4
#define SEC_CHUNK (KCHUNKS / SECTIONS)    // 9 chunks (= 32 input cols) / section
#define ROWS_WG   128
#define THREADS   128                     // 4 waves, 2 M-tiles each
#define A_BYTES   (SEC_CHUNK * 8 * 32 * 32)   // 9 chunks * 8 Mtiles * 1KB = 73728
#define B_BYTES   (SEC_CHUNK * 8 * 32 * 32)   // 9 chunks * 8 Ntiles * 1KB = 73728
#define LDS_BYTES (A_BYTES + B_BYTES)         // 147456 -> 2 WGs / WGP

__device__ __forceinline__ unsigned short f2bf(float f) {
    unsigned int u = __float_as_uint(f);
    u += 0x7FFFu + ((u >> 16) & 1u);      // round to nearest even
    return (unsigned short)(u >> 16);
}

// ---------------------------------------------------------------------------
// Pack jacobi_coeffs (i, o, d) fp32 -> bf16 B-fragments in d_ws.
// B fragment layout (32x16 bf16, 8 VGPRs/lane): lanes 0-15 hold K=0-7,16-23,
// lanes 16-31 hold K=8-15,24-31; lane%16 = N; VGPR v half h <-> klocal = 8g+2v+h (v<4), 16+8g+2(v-4)+h (v>=4).
// Element index: ((kc*8 + nt)*32 + lane)*16 + slot,  slot = 2v+h.
// ---------------------------------------------------------------------------
__global__ void pack_coeffs(const float* __restrict__ coeffs,
                            unsigned short* __restrict__ wsB) {
    int t = blockIdx.x * blockDim.x + threadIdx.x;   // over 36*8*512 elements
    if (t >= KCHUNKS * 8 * 512) return;
    int slot = t & 15;
    int lane = (t >> 4) & 31;
    int nt   = (t >> 9) & 7;
    int kc   = t >> 12;
    int klocal = (slot < 8) ? (((lane >> 4) << 3) + slot) : (16 + ((lane >> 4) << 3) + (slot - 8));
    int k = kc * 32 + klocal;                        // k = i*9 + d
    int i = k / 9;
    int d = k - i * 9;
    int o = nt * 16 + (lane & 15);
    wsB[t] = f2bf(coeffs[(i * OUT_DIM + o) * NDEG + d]);
}

// ---------------------------------------------------------------------------
// Main kernel: one workgroup = 128 batch rows, 4 waves; wave w owns M-tiles
// {2w, 2w+1} (rows [32w, 32w+32)) and the full 128-wide output (8 N-tiles).
// ---------------------------------------------------------------------------
__global__ void __launch_bounds__(THREADS)
kan_main(const float* __restrict__ x,
         const unsigned short* __restrict__ wsBu,
         float* __restrict__ out) {
    extern __shared__ unsigned char lds[];   // [0, A_BYTES): A, [A_BYTES, +B_BYTES): B
    const int tid  = threadIdx.x;
    const int lane = tid & 31;
    const int wave = tid >> 5;                       // 0..3
    const long wgRow = (long)blockIdx.x * ROWS_WG;

    v8f acc0[8] = {};                                // M-tile 2w
    v8f acc1[8] = {};                                // M-tile 2w+1

    for (int s = 0; s < SECTIONS; ++s) {
        __syncthreads();                             // LDS panel reuse (WAR)

        // ---- Phase A0: copy this section's B fragments global -> LDS,
        //      once per workgroup (36 b128 per thread). Loads issued here
        //      so their latency overlaps the tanh/recurrence VALU below.
        {
            const uint4* src = reinterpret_cast<const uint4*>(
                wsBu + (size_t)s * (B_BYTES / 2));
            uint4* dst = reinterpret_cast<uint4*>(lds + A_BYTES);
            #pragma unroll 4
            for (int j = tid; j < B_BYTES / 16; j += THREADS) {
                dst[j] = src[j];
            }
        }

        // ---- Phase A1: tanh + Jacobi recurrence -> bf16 A-fragments in LDS.
        // 128 rows x 32 input cols per section; 32 elements per thread.
        {
            const int il = tid & 31;                 // input col within section
            const int r0 = tid >> 5;                 // 0..3
            for (int rr = 0; rr < 32; ++rr) {
                const int row = r0 + rr * 4;         // 0..127
                const float xv = x[(wgRow + row) * IN_DIM + s * 32 + il];
                const float tx = tanhf(xv);
                const int mt = row >> 4;
                const int m  = row & 15;

                // scatter one basis value into the A-fragment LDS layout:
                // 16-bit A 16x32: lane = g*16 + m; VGPR v half h <-> klocal:
                //   v<4 : klocal = 8g + 2v+h ;  v>=4 : klocal = 16 + 8g + 2(v-4)+h
                auto stash = [&](int d, float val) {
                    int ks = il * 9 + d;             // k within section (0..287)
                    int kc = ks >> 5;                // chunk in section (0..8)
                    int kl = ks & 31;
                    int g, r2, v;
                    if (kl < 16) { g = kl >> 3; r2 = kl & 7; v = r2 >> 1; }
                    else { int q = kl - 16; g = q >> 3; r2 = q & 7; v = 4 + (r2 >> 1); }
                    int h = r2 & 1;
                    int laneW = g * 16 + m;
                    int off = ((kc * 8 + mt) * 32 + laneW) * 32 + (v * 2 + h) * 2;
                    *reinterpret_cast<unsigned short*>(lds + off) = f2bf(val);
                };

                float pm2 = 1.0f;                    // P0^(1,1)(t) = 1
                float pm1 = 2.0f * tx;               // P1^(1,1)(t) = 2t (a=b=1)
                stash(0, pm2);
                stash(1, pm1);
                #pragma unroll
                for (int n = 2; n <= 8; ++n) {       // a=b=1 => B-coeff = 0
                    const float An = (float)((2 * n + 1) * (n + 1)) /
                                     (float)(n * (n + 2));
                    const float Cn = -(float)(n + 1) / (float)(n + 2);
                    float p = fmaf(An * tx, pm1, Cn * pm2);
                    stash(n, p);
                    pm2 = pm1; pm1 = p;
                }
            }
        }
        __syncthreads();

        // ---- Phase B: 9 chunks; per chunk 2 A-frags (held in regs) and a
        //      2-deep rotated stream of 8 B-frags from LDS feeding 16 WMMAs.
        auto loadB = [&](int kc, int nt) -> v16bf {
            return *reinterpret_cast<const v16bf*>(
                lds + A_BYTES + ((kc * 8 + nt) * 32 + lane) * 32);
        };
        #pragma unroll
        for (int kc = 0; kc < SEC_CHUNK; ++kc) {
            const v16bf a0 = *reinterpret_cast<const v16bf*>(
                lds + ((kc * 8 + 2 * wave + 0) * 32 + lane) * 32);
            const v16bf a1 = *reinterpret_cast<const v16bf*>(
                lds + ((kc * 8 + 2 * wave + 1) * 32 + lane) * 32);

            v16bf bc = loadB(kc, 0);
            #pragma unroll
            for (int nt = 0; nt < 8; ++nt) {
                v16bf bn = loadB(kc, (nt + 1) & 7);  // prefetch next (wraps)
                acc0[nt] = __builtin_amdgcn_wmma_f32_16x16x32_bf16(
                    false, a0, false, bc, (short)0, acc0[nt], false, false);
                acc1[nt] = __builtin_amdgcn_wmma_f32_16x16x32_bf16(
                    false, a1, false, bc, (short)0, acc1[nt], false, false);
                bc = bn;
            }
        }
    }

    // ---- Store C: 16x16 f32 D layout: lanes 0-15: M=r, N=lane;
    //               lanes 16-31: M=r+8, N=lane-16.
    const int nOff = lane & 15;
    const int mOff = (lane >> 4) << 3;
    #pragma unroll
    for (int nt = 0; nt < 8; ++nt) {
        #pragma unroll
        for (int r = 0; r < 8; ++r) {
            const long rb = wgRow + wave * 32 + mOff + r;
            out[(rb +  0) * OUT_DIM + nt * 16 + nOff] = acc0[nt][r];
            out[(rb + 16) * OUT_DIM + nt * 16 + nOff] = acc1[nt][r];
        }
    }
}

// ---------------------------------------------------------------------------
extern "C" void kernel_launch(void* const* d_in, const int* in_sizes, int n_in,
                              void* d_out, int out_size, void* d_ws, size_t ws_size,
                              hipStream_t stream) {
    const float* x      = (const float*)d_in[0];
    const float* coeffs = (const float*)d_in[1];
    float* out          = (float*)d_out;
    unsigned short* wsB = (unsigned short*)d_ws;     // 294912 bytes of bf16 B

    const int packN = KCHUNKS * 8 * 512;             // 147456 bf16 elements
    pack_coeffs<<<packN / 256, 256, 0, stream>>>(coeffs, wsB);

    const int rows = in_sizes[0] / IN_DIM;           // 262144
    kan_main<<<rows / ROWS_WG, THREADS, LDS_BYTES, stream>>>(x, wsB, out);
}
